// LlamaAttentionExperimental_66726611911316
// MI455X (gfx1250) — matrix operand order repaired and checked by reading, not compiled, not run on hardware
//
#include <hip/hip_runtime.h>

// ---------------- problem constants ----------------
constexpr int S_LEN  = 2048;
constexpr int HID    = 4096;
constexpr int NH     = 32;
constexpr int KVH    = 8;
constexpr int HD     = 128;           // head dim
constexpr int QDIM   = NH * HD;       // 4096
constexpr int KVDIM  = KVH * HD;      // 1024
constexpr int NKEEP  = 1024;          // int(S * 0.5)
constexpr int SINK   = 4;
constexpr float NEG_INF = -3.402823466e38f;
constexpr float SCORE_SCALE = 0.08838834764831845f;  // 1/sqrt(128)

typedef __attribute__((ext_vector_type(16))) __bf16 v16bf;
typedef __attribute__((ext_vector_type(8)))  float  v8f;
typedef __attribute__((ext_vector_type(4)))  unsigned int u32x4;
typedef __attribute__((ext_vector_type(4)))  int          i32x4;
typedef __attribute__((ext_vector_type(8)))  int          i32x8;

// ---------------- WMMA core ----------------
__device__ inline v8f wmma_bf16(v16bf a, v16bf b, v8f c) {
  return __builtin_amdgcn_wmma_f32_16x16x32_bf16(
      false, a, false, b, (short)0, c, false, false);
}

// A fragment (16x32 bf16) from row-major LDS tile.
__device__ inline v16bf lds_a_frag(const __bf16* t, int ld, int lane) {
  int r = lane & 15, hi = lane >> 4;
  const __bf16* p = t + r * ld + hi * 16;
  v16bf a;
#pragma unroll
  for (int j = 0; j < 8; ++j) { a[j] = p[j]; a[8 + j] = p[8 + j]; }
  return a;
}

// B fragment (32x16) from column-major LDS tile ([N][32] bf16):
// per lane: 16 contiguous K values -> two ds_load_b128.
__device__ inline v16bf lds_b_frag_cm(const __bf16* colbase, int lane) {
  int c = lane & 15, kh = lane >> 4;
  const __bf16* p = colbase + c * 32 + kh * 16;
  union { uint4 q[2]; v16bf v; } u;
  u.q[0] = *(const uint4*)p;
  u.q[1] = *(const uint4*)(p + 8);
  return u.v;
}

// B fragment when the 16 K-values of this lane are contiguous in memory.
__device__ inline v16bf g_b_frag_contig(const __bf16* p) {
  union { uint4 q[2]; v16bf v; } u;
  u.q[0] = ((const uint4*)p)[0];
  u.q[1] = ((const uint4*)p)[1];
  return u.v;
}

// ---------------- wave32 helpers ----------------
__device__ inline int wred_sum_i(int v) {
#pragma unroll
  for (int o = 16; o > 0; o >>= 1) v += __shfl_xor(v, o, 32);
  return v;
}
__device__ inline float wred_sum_f(float v) {
#pragma unroll
  for (int o = 16; o > 0; o >>= 1) v += __shfl_xor(v, o, 32);
  return v;
}
__device__ inline float wred_max_f(float v) {
#pragma unroll
  for (int o = 16; o > 0; o >>= 1) v = fmaxf(v, __shfl_xor(v, o, 32));
  return v;
}
// uniform count of a wave-wide predicate (v_cmp -> mask, then scalar popcount)
__device__ inline int wave_count(bool pred) {
#if __has_builtin(__builtin_amdgcn_ballot_w32)
  return __builtin_popcount(__builtin_amdgcn_ballot_w32(pred));
#else
  return wred_sum_i(pred ? 1 : 0);
#endif
}

// monotone f32 <-> u32 order-preserving map (for exact top-k)
__device__ inline unsigned f2ord(float f) {
  unsigned b = __float_as_uint(f);
  return b ^ ((b & 0x80000000u) ? 0xFFFFFFFFu : 0x80000000u);
}
__device__ inline float ord2f(unsigned u) {
  unsigned b = (u & 0x80000000u) ? (u ^ 0x80000000u) : ~u;
  return __uint_as_float(b);
}

// async global -> LDS staging (ASYNCcnt path)
__device__ inline void async_g2l_b128(unsigned lds_byte_off, const void* gptr) {
  unsigned long long ga = (unsigned long long)(size_t)gptr;
  asm volatile("global_load_async_to_lds_b128 %0, %1, off"
               :: "v"(lds_byte_off), "v"(ga) : "memory");
}
__device__ inline void wait_asynccnt0() {
#if __has_builtin(__builtin_amdgcn_s_wait_asynccnt)
  __builtin_amdgcn_s_wait_asynccnt(0);
#else
  asm volatile("s_wait_asynccnt 0x0" ::: "memory");
#endif
}

// ---------------- elementwise kernels ----------------
__global__ void cvt_f32_to_bf16(const float* __restrict__ x,
                                __bf16* __restrict__ y, int n) {
  for (int i = blockIdx.x * blockDim.x + threadIdx.x; i < n;
       i += gridDim.x * blockDim.x)
    y[i] = (__bf16)x[i];
}

// V transpose: Vt[(kvh*128+d)*S + s] = V[s*1024 + kvh*128 + d]
__global__ void transpose_v(const __bf16* __restrict__ V,
                            __bf16* __restrict__ Vt) {
  for (int i = blockIdx.x * blockDim.x + threadIdx.x; i < S_LEN * KVDIM;
       i += gridDim.x * blockDim.x) {
    int s = i >> 10, cd = i & (KVDIM - 1);
    Vt[(size_t)cd * S_LEN + s] = V[i];
  }
}

// In-place RoPE on bf16 [S, heads*128]; thread owns the (d, d+64) pair.
__global__ void rope_bf16(__bf16* __restrict__ X, const float* __restrict__ cosT,
                          const float* __restrict__ sinT, int heads) {
  int idx = blockIdx.x * blockDim.x + threadIdx.x;
  int total = S_LEN * heads * 64;
  if (idx >= total) return;
  int d = idx & 63;
  int h = (idx >> 6) % heads;
  int s = idx / (heads * 64);
  __bf16* row = X + (size_t)s * heads * HD + h * HD;
  float x1 = (float)row[d], x2 = (float)row[d + 64];
  float c1 = cosT[s * HD + d],      s1 = sinT[s * HD + d];
  float c2 = cosT[s * HD + d + 64], s2 = sinT[s * HD + d + 64];
  row[d]      = (__bf16)(x1 * c1 - x2 * s1);
  row[d + 64] = (__bf16)(x2 * c2 + x1 * s2);
}

// ---------------- generic WMMA GEMM: C[M,N] = A[M,K] * B[K,N] ----------------
// 128x128 block tile, BK=32, 256 threads (8 waves), wave = 64x32 (4x2 WMMA).
// Double-buffered LDS pipeline: stage kt+1 via global_load_async_to_lds_b128
// (A, ASYNCcnt) + transpose-on-store (B) while WMMAs consume stage kt.
template <typename OutT>
__global__ __launch_bounds__(256) void gemm_wmma_bf16(
    const __bf16* __restrict__ A, const __bf16* __restrict__ B,
    OutT* __restrict__ C, int M, int N, int K) {
  __shared__ __bf16 As[2][128 * 32];   // row-major [128][32]
  __shared__ __bf16 Bs[2][128 * 32];   // column-major [128 n][32 k]

  const int tid = threadIdx.x;
  const int lane = tid & 31, w = tid >> 5;
  const int mBase = blockIdx.y * 128;
  const int nBase = blockIdx.x * 128;
  const int mblk = (w & 1) * 64;
  const int nblk = (w >> 1) * 32;
  const int KT = K >> 5;
  const unsigned asBase = (unsigned)(size_t)(void*)&As[0][0];

  v8f acc[4][2] = {};

  auto stage = [&](int kt, int buf) {
    // A tile: async DMA global -> LDS, 16B per lane
#pragma unroll
    for (int i = 0; i < 2; ++i) {
      int uid = tid * 2 + i;
      int r = uid >> 2, c8 = (uid & 3) * 8;
      async_g2l_b128(asBase + (unsigned)(buf * 8192 + (r * 32 + c8) * 2),
                     A + (size_t)(mBase + r) * K + kt * 32 + c8);
    }
    // B tile: coalesced load, transpose-on-store to column-major
#pragma unroll
    for (int i = 0; i < 2; ++i) {
      int uid = tid * 2 + i;
      int r = uid >> 4, c8 = (uid & 15) * 8;
      union { uint4 q; __bf16 e[8]; } ub;
      ub.q = *(const uint4*)(B + (size_t)(kt * 32 + r) * N + nBase + c8);
#pragma unroll
      for (int e2 = 0; e2 < 8; ++e2) Bs[buf][(c8 + e2) * 32 + r] = ub.e[e2];
    }
    if (kt + 1 < KT) {  // global_prefetch_b8 the tile after next
      int uid = tid * 2;
      __builtin_prefetch(A + (size_t)(mBase + (uid >> 2)) * K + (kt + 1) * 32, 0, 0);
      __builtin_prefetch(B + (size_t)((kt + 1) * 32 + (uid >> 4)) * N + nBase, 0, 0);
    }
  };

  stage(0, 0);
  wait_asynccnt0();
  __syncthreads();

  for (int kt = 0; kt < KT; ++kt) {
    const int cur = kt & 1;
    if (kt + 1 < KT) stage(kt + 1, cur ^ 1);   // overlap with compute below

    v16bf aF[4], bF[2];
#pragma unroll
    for (int mi = 0; mi < 4; ++mi)
      aF[mi] = lds_a_frag(As[cur] + (mblk + mi * 16) * 32, 32, lane);
#pragma unroll
    for (int ni = 0; ni < 2; ++ni)
      bF[ni] = lds_b_frag_cm(Bs[cur] + (nblk + ni * 16) * 32, lane);
#pragma unroll
    for (int mi = 0; mi < 4; ++mi)
#pragma unroll
      for (int ni = 0; ni < 2; ++ni)
        acc[mi][ni] = wmma_bf16(aF[mi], bF[ni], acc[mi][ni]);

    wait_asynccnt0();   // next buffer's async transfers landed
    __syncthreads();    // + everyone done reading current buffer
  }

  const int c = lane & 15, hi = lane >> 4;
#pragma unroll
  for (int mi = 0; mi < 4; ++mi)
#pragma unroll
    for (int ni = 0; ni < 2; ++ni)
#pragma unroll
      for (int i = 0; i < 8; ++i) {
        int m = mBase + mblk + mi * 16 + i + hi * 8;
        int n = nBase + nblk + ni * 16 + c;
        C[(size_t)m * N + n] = (OutT)acc[mi][ni][i];
      }
}

// ---------------- sparse attention kernel ----------------
// One WG per (head, 32-query tile); 32 waves; 32x2048 f32 scores resident in
// LDS (256 KB of 320 KB) -> exact per-row top-1024 in one pass.
constexpr int ATTN_SMEM = 32 * S_LEN * 4 + 32 * HD * 2 + 128;

__global__ __launch_bounds__(1024, 1) void attn_sparse_topk(
    const __bf16* __restrict__ Qb, const __bf16* __restrict__ Kb,
    const __bf16* __restrict__ Vt, __bf16* __restrict__ AOb) {
  extern __shared__ __align__(16) char smem[];
  float*  sc   = (float*)smem;                               // [32][2048]
  __bf16* qs   = (__bf16*)(smem + 32 * S_LEN * 4);           // [32][128]
  float*  rsum = (float*)(smem + 32 * S_LEN * 4 + 32 * HD * 2);

  const int h   = blockIdx.y;
  const int q0  = blockIdx.x * 32;
  const int kvh = h >> 2;  // GROUPS = 4
  const int tid = threadIdx.x;
  const int lane = tid & 31, w = tid >> 5;

  // ---- phase 0: stage rope'd Q tile via the Tensor Data Mover ----
#if __has_builtin(__builtin_amdgcn_tensor_load_to_lds)
  if (tid < 32) {
    unsigned lds_off = (unsigned)(size_t)(void*)qs;
    unsigned long long ga =
        (unsigned long long)(size_t)(Qb + (size_t)q0 * QDIM + h * HD);
    u32x4 g0;
    g0[0] = 1u;                                   // count=1 descriptor
    g0[1] = lds_off;                              // lds_addr
    g0[2] = (unsigned)ga;                         // global_addr[31:0]
    g0[3] = ((unsigned)(ga >> 32) & 0x01FFFFFFu)  // global_addr[56:32]
            | (2u << 30);                         // type = 2 ("image")
    i32x8 g1;
    g1[0] = (int)(1u << 16);        // data_size = 2 bytes, no multicast
    g1[1] = 0;                      // no atomic barrier; tensor_dim0 lo16 = 0
    g1[2] = 0x10;                   // tensor_dim0 = 1<<20; tensor_dim1 lo16 = 0
    g1[3] = (int)(0x10u | (128u << 16)); // tensor_dim1 = 1<<20; tile_dim0 = 128
    g1[4] = 32;                     // tile_dim1 = 32 rows; tile_dim2 = 0
    g1[5] = QDIM;                   // tensor_dim0_stride = 4096 elements
    g1[6] = 0;
    g1[7] = 0;
    i32x4 z4 = {};
#if defined(__clang_major__) && __clang_major__ >= 23
    i32x8 z8 = {};
    __builtin_amdgcn_tensor_load_to_lds(g0, g1, z4, z4, z8, 0);
#else
    __builtin_amdgcn_tensor_load_to_lds(g0, g1, z4, z4, 0);
#endif
    __builtin_amdgcn_s_wait_tensorcnt(0);
  }
#else
#pragma unroll
  for (int i = 0; i < 4; ++i) {
    int e = tid * 4 + i;
    int r = e >> 7, cc = e & 127;
    qs[e] = Qb[(size_t)(q0 + r) * QDIM + h * HD + cc];
  }
#endif
  __syncthreads();

  // ---- phase 1: scores = Q * K^T (WMMA), scaled + causal-masked -> LDS ----
  {
    const int mt = w & 1;
    const int ng = w >> 1;
    const int c = lane & 15, kh = lane >> 4;
    v16bf aF[4];
#pragma unroll
    for (int ks = 0; ks < 4; ++ks)
      aF[ks] = lds_a_frag(qs + mt * 16 * HD + ks * 32, HD, lane);
    for (int nt0 = 0; nt0 < 8; ++nt0) {
      int ntile = ng * 8 + nt0;
      int key = ntile * 16 + c;
      v8f acc = {};
#pragma unroll
      for (int ks = 0; ks < 4; ++ks) {
        v16bf bF = g_b_frag_contig(
            Kb + (size_t)key * KVDIM + kvh * HD + ks * 32 + kh * 16);
        acc = wmma_bf16(aF[ks], bF, acc);
      }
#pragma unroll
      for (int i = 0; i < 8; ++i) {
        int ml = mt * 16 + i + kh * 8;
        float v = acc[i] * SCORE_SCALE;
        if (key > q0 + ml) v = NEG_INF;
        sc[ml * S_LEN + key] = v;
      }
    }
  }
  __syncthreads();

  // ---- phase 2: per-row exact top-k threshold + softmax (wave w = row w) ----
  {
    const int r = w;
    const int qg = q0 + r;
    unsigned u[64];
#pragma unroll
    for (int t = 0; t < 64; ++t)
      u[t] = f2ord(sc[r * S_LEN + t * 32 + lane]);

    unsigned T = 0u;
    if (qg + 1 > NKEEP) {
      for (int bit = 31; bit >= 0; --bit) {
        unsigned cand = T | (1u << bit);
        int cnt = 0;
#pragma unroll
        for (int t = 0; t < 64; ++t) cnt += wave_count(u[t] >= cand);
        if (cnt >= NKEEP) T = cand;
      }
    }

    float mx = NEG_INF;
#pragma unroll
    for (int t = 0; t < 64; ++t) {
      int e = t * 32 + lane;
      bool keep = (e <= qg) && ((u[t] >= T) || (e < SINK));
      if (keep) mx = fmaxf(mx, ord2f(u[t]));
    }
    mx = wred_max_f(mx);

    float sum = 0.f;
#pragma unroll
    for (int t = 0; t < 64; ++t) {
      int e = t * 32 + lane;
      bool keep = (e <= qg) && ((u[t] >= T) || (e < SINK));
      float p = keep ? __expf(ord2f(u[t]) - mx) : 0.f;
      sum += p;
      sc[r * S_LEN + e] = p;   // unnormalized P, in place
    }
    sum = wred_sum_f(sum);
    if (lane == 0) rsum[r] = sum;
  }
  __syncthreads();

  // ---- phase 3: out = P * V (WMMA); V pre-transposed -> contiguous B frags --
  if (w < 16) {
    const int mt = w & 1;
    const int nt = w >> 1;
    const int c = lane & 15, kh = lane >> 4;
    const int dcol = nt * 16 + c;
    const __bf16* vrow = Vt + (size_t)(kvh * HD + dcol) * S_LEN;
    v8f acc = {};
    for (int ks = 0; ks < S_LEN / 32; ++ks) {
      v16bf aF;
      {
        int rr = lane & 15;
        const float* p = sc + (mt * 16 + rr) * S_LEN + ks * 32 + kh * 16;
#pragma unroll
        for (int j = 0; j < 8; ++j) {
          aF[j] = (__bf16)p[j];
          aF[8 + j] = (__bf16)p[8 + j];
        }
      }
      v16bf bF = g_b_frag_contig(vrow + ks * 32 + kh * 16);
      acc = wmma_bf16(aF, bF, acc);
    }
#pragma unroll
    for (int i = 0; i < 8; ++i) {
      int ml = mt * 16 + i + kh * 8;
      float o = acc[i] * (1.0f / rsum[ml]);
      AOb[(size_t)(q0 + ml) * QDIM + h * HD + dcol] = (__bf16)o;
    }
  }
}

// ---------------- host side ----------------
extern "C" void kernel_launch(void* const* d_in, const int* in_sizes, int n_in,
                              void* d_out, int out_size, void* d_ws, size_t ws_size,
                              hipStream_t stream) {
  const float* hs   = (const float*)d_in[0];
  const float* Wq   = (const float*)d_in[1];
  const float* Wk   = (const float*)d_in[2];
  const float* Wv   = (const float*)d_in[3];
  const float* Wo   = (const float*)d_in[4];
  const float* cosT = (const float*)d_in[5];
  const float* sinT = (const float*)d_in[6];
  float* out = (float*)d_out;

  char* ws = (char*)d_ws;
  size_t off = 0;
  auto alloc = [&](size_t bytes) -> void* {
    void* p = ws + off;
    off += (bytes + 255) & ~(size_t)255;
    return p;
  };
  __bf16* Hb  = (__bf16*)alloc((size_t)S_LEN * HID * 2);
  __bf16* Wqb = (__bf16*)alloc((size_t)HID * QDIM * 2);
  __bf16* Wkb = (__bf16*)alloc((size_t)HID * KVDIM * 2);
  __bf16* Wvb = (__bf16*)alloc((size_t)HID * KVDIM * 2);
  __bf16* Wob = (__bf16*)alloc((size_t)QDIM * HID * 2);
  __bf16* Qb  = (__bf16*)alloc((size_t)S_LEN * QDIM * 2);
  __bf16* Kb  = (__bf16*)alloc((size_t)S_LEN * KVDIM * 2);
  __bf16* Vb  = (__bf16*)alloc((size_t)S_LEN * KVDIM * 2);
  __bf16* Vt  = (__bf16*)alloc((size_t)S_LEN * KVDIM * 2);
  __bf16* AOb = (__bf16*)alloc((size_t)S_LEN * QDIM * 2);

  cvt_f32_to_bf16<<<4096, 256, 0, stream>>>(hs, Hb, S_LEN * HID);
  cvt_f32_to_bf16<<<4096, 256, 0, stream>>>(Wq, Wqb, HID * QDIM);
  cvt_f32_to_bf16<<<4096, 256, 0, stream>>>(Wk, Wkb, HID * KVDIM);
  cvt_f32_to_bf16<<<4096, 256, 0, stream>>>(Wv, Wvb, HID * KVDIM);
  cvt_f32_to_bf16<<<4096, 256, 0, stream>>>(Wo, Wob, QDIM * HID);

  gemm_wmma_bf16<__bf16><<<dim3(QDIM / 128, S_LEN / 128), 256, 0, stream>>>(
      Hb, Wqb, Qb, S_LEN, QDIM, HID);
  gemm_wmma_bf16<__bf16><<<dim3(KVDIM / 128, S_LEN / 128), 256, 0, stream>>>(
      Hb, Wkb, Kb, S_LEN, KVDIM, HID);
  gemm_wmma_bf16<__bf16><<<dim3(KVDIM / 128, S_LEN / 128), 256, 0, stream>>>(
      Hb, Wvb, Vb, S_LEN, KVDIM, HID);

  rope_bf16<<<(S_LEN * NH * 64) / 256, 256, 0, stream>>>(Qb, cosT, sinT, NH);
  rope_bf16<<<(S_LEN * KVH * 64) / 256, 256, 0, stream>>>(Kb, cosT, sinT, KVH);
  transpose_v<<<4096, 256, 0, stream>>>(Vb, Vt);

  attn_sparse_topk<<<dim3(S_LEN / 32, NH), 1024, ATTN_SMEM, stream>>>(
      Qb, Kb, Vt, AOb);

  gemm_wmma_bf16<float><<<dim3(HID / 128, S_LEN / 128), 256, 0, stream>>>(
      AOb, Wob, out, S_LEN, HID, QDIM);
}
